// CheemsMambaMixer_17592186044954
// MI455X (gfx1250) — matrix operand order repaired and checked by reading, not compiled, not run on hardware
//
#include <hip/hip_runtime.h>
#include <hip/hip_fp16.h>

typedef _Float16 h16;
typedef __attribute__((ext_vector_type(16))) _Float16 v16h;
typedef __attribute__((ext_vector_type(8)))  _Float16 v8h;
typedef __attribute__((ext_vector_type(8)))  float    v8f;

#define LL   2048
#define HH   1024
#define DI   2048
#define DS   16
#define DTR  64
#define NSSM 96   // DTR + 2*DS

// ---------------------------------------------------------------------------
// Explicit global-address-space loads (force global_load_b128, never flat).
// Global AS1 pointers share the 64-bit representation of generic pointers to
// global memory on amdgcn, so reconstructing via the integer value is exact.
// ---------------------------------------------------------------------------
__device__ __forceinline__ v16h gload16(const h16* p) {
  return *reinterpret_cast<const __attribute__((address_space(1))) v16h*>(
      (unsigned long long)p);
}
__device__ __forceinline__ v8h gload8(const h16* p) {
  return *reinterpret_cast<const __attribute__((address_space(1))) v8h*>(
      (unsigned long long)p);
}

// ---------------------------------------------------------------------------
// Elementwise helpers
// ---------------------------------------------------------------------------
__global__ void k_f32_to_f16(const float* __restrict__ src, h16* __restrict__ dst, int n) {
  int i = blockIdx.x * blockDim.x + threadIdx.x;
  if (i < n) dst[i] = (h16)src[i];
}

__global__ void k_negexp(const float* __restrict__ src, float* __restrict__ dst, int n) {
  int i = blockIdx.x * blockDim.x + threadIdx.x;
  if (i < n) dst[i] = -__expf(src[i]);
}

// ---------------------------------------------------------------------------
// WMMA GEMM:  C[M,N] (f32) = A[M,K] (f16, row-major) * W[N,K]^T (f16, row-major)
// Each wave computes a 32 x (16*NT) tile: 2 A-frags + NT B-frags -> 2*NT WMMAs
// per K-step, B reused across both M-subtiles. Block = 4 waves stacked in M
// (128 rows). Requirements: M % 128 == 0, K % 64 == 0, N % (16*NT) == 0.
// Loads grouped before the WMMA chain so waits can be partial.
// ---------------------------------------------------------------------------
template <int NT>
__global__ __launch_bounds__(128)
void k_gemm_f16(const h16* __restrict__ A, const h16* __restrict__ W,
                float* __restrict__ C, int N, int K) {
  const int lane = threadIdx.x & 31;
  const int wave = threadIdx.x >> 5;
  const int m0   = blockIdx.x * 128 + wave * 32;
  const int nb   = blockIdx.y * (16 * NT);
  const int half = lane >> 4;   // which K-half of the fragment this lane holds
  const int sub  = lane & 15;

  // A 16x32 layout: lane holds row m0+sub, K = k0+16*half..+15 (contiguous, matches B)
  const h16* pa0 = A + (size_t)(m0 + sub) * K + (half << 4);
  const h16* pa1 = pa0 + (size_t)16 * K;
  // B 32x16 (KxN): lane holds col nb+t*16+sub, K = k0+16*half..+15 (contiguous W row)
  const h16* pb[NT];
#pragma unroll
  for (int t = 0; t < NT; ++t)
    pb[t] = W + (size_t)(nb + t * 16 + sub) * K + (half << 4);

  v8f acc0[NT], acc1[NT];
#pragma unroll
  for (int t = 0; t < NT; ++t) { acc0[t] = v8f{}; acc1[t] = v8f{}; }

#pragma unroll 2
  for (int k0 = 0; k0 < K; k0 += 32) {
    // stream prefetch ~2 K-iterations ahead (global_prefetch_b8)
    __builtin_prefetch(pa0 + 64, 0, 1);
    __builtin_prefetch(pb[0] + 64, 0, 1);

    // ---- load group ----
    v8h a0lo = gload8(pa0), a0hi = gload8(pa0 + 8);
    v8h a1lo = gload8(pa1), a1hi = gload8(pa1 + 8);
    v16h b[NT];
#pragma unroll
    for (int t = 0; t < NT; ++t) { b[t] = gload16(pb[t]); pb[t] += 32; }
    pa0 += 32; pa1 += 32;

    v16h a0, a1;
#pragma unroll
    for (int i = 0; i < 8; ++i) {
      a0[i] = a0lo[i]; a0[i + 8] = a0hi[i];
      a1[i] = a1lo[i]; a1[i + 8] = a1hi[i];
    }

    // ---- WMMA chain ----
#pragma unroll
    for (int t = 0; t < NT; ++t) {
      acc0[t] = __builtin_amdgcn_wmma_f32_16x16x32_f16(
          false, a0, false, b[t], (short)0, acc0[t], false, false);
      acc1[t] = __builtin_amdgcn_wmma_f32_16x16x32_f16(
          false, a1, false, b[t], (short)0, acc1[t], false, false);
    }
  }

  // D layout: VGPR r -> row (m-subtile base) + r + 8*half, col nb + t*16 + sub
  const int mb0 = m0 + (half << 3);
#pragma unroll
  for (int t = 0; t < NT; ++t) {
    float* cp0 = C + (size_t)mb0 * N + nb + t * 16 + sub;
    float* cp1 = cp0 + (size_t)16 * N;
#pragma unroll
    for (int r = 0; r < 8; ++r) {
      cp0[(size_t)r * N] = acc0[t][r];
      cp1[(size_t)r * N] = acc1[t][r];
    }
  }
}

// ---------------------------------------------------------------------------
// Causal depthwise conv1d (K=4) + bias + SiLU. x = proj[:, 0:DI].
// ---------------------------------------------------------------------------
__global__ void k_conv_silu(const float* __restrict__ proj, const float* __restrict__ cw,
                            const float* __restrict__ cb, float* __restrict__ xcf,
                            h16* __restrict__ xch) {
  int i = blockIdx.x * blockDim.x + threadIdx.x;  // i = l*DI + d
  if (i >= LL * DI) return;
  int d = i & (DI - 1);
  int l = i >> 11;
  float s = cb[d];
#pragma unroll
  for (int j = 0; j < 4; ++j) {
    int lj = l - 3 + j;
    if (lj >= 0) s += cw[d * 4 + j] * proj[(size_t)lj * (2 * DI) + d];
  }
  float v = s / (1.f + __expf(-s));               // SiLU
  xcf[i] = v;
  xch[i] = (h16)v;
}

// dt input (first DTR cols of ssm) -> f16 for the dt_proj GEMM
__global__ void k_split_dt(const float* __restrict__ ssm, h16* __restrict__ dtin) {
  int i = blockIdx.x * blockDim.x + threadIdx.x;  // i = l*DTR + t
  if (i >= LL * DTR) return;
  int l = i >> 6, t = i & 63;
  dtin[i] = (h16)ssm[l * NSSM + t];
}

// dt = softplus(dt_raw + bias), in place
__global__ void k_dt_softplus(float* __restrict__ dt, const float* __restrict__ bias) {
  int i = blockIdx.x * blockDim.x + threadIdx.x;
  if (i >= LL * DI) return;
  int d = i & (DI - 1);
  float x = dt[i] + bias[d];
  dt[i] = (x > 20.f) ? x : log1pf(__expf(x));
}

// ---------------------------------------------------------------------------
// Selective scan. One lane per (channel d, state n): 16 lanes share a channel.
// h_l = exp(dt*A)*h_{l-1} + dt*B_l*xc ; y = sum_n h*C_l via 16-lane xor tree.
// ---------------------------------------------------------------------------
__global__ void k_scan(const float* __restrict__ dt, const float* __restrict__ xcf,
                       const float* __restrict__ ssm, const float* __restrict__ Aneg,
                       float* __restrict__ y) {
  int tid  = blockIdx.x * blockDim.x + threadIdx.x;
  int lane = tid & 31;
  int n    = lane & 15;
  int d    = ((tid >> 5) << 1) + (lane >> 4);
  float a = Aneg[d * DS + n];
  float h = 0.f;
  for (int l = 0; l < LL; ++l) {
    const float* bc = ssm + (size_t)l * NSSM;
    float dtv = dt[(size_t)l * DI + d];
    float xv  = xcf[(size_t)l * DI + d];
    float dA  = __expf(dtv * a);
    h = dA * h + (dtv * xv) * bc[DTR + n];        // B at cols 64..79
    float p = h * bc[DTR + DS + n];               // C at cols 80..95
#pragma unroll
    for (int off = 1; off < 16; off <<= 1)
      p += __shfl_xor(p, off, 16);
    if (n == 0) y[(size_t)l * DI + d] = p;
  }
}

// yg = (y + xc*D) * silu(z), z = proj[:, DI:2*DI]; output f16 for out_proj GEMM
__global__ void k_gate(const float* __restrict__ y, const float* __restrict__ xcf,
                       const float* __restrict__ Dv, const float* __restrict__ proj,
                       h16* __restrict__ yg) {
  int i = blockIdx.x * blockDim.x + threadIdx.x;
  if (i >= LL * DI) return;
  int d = i & (DI - 1);
  int l = i >> 11;
  float z = proj[(size_t)l * (2 * DI) + DI + d];
  float g = z / (1.f + __expf(-z));
  yg[i] = (h16)((y[i] + xcf[i] * Dv[d]) * g);
}

// ---------------------------------------------------------------------------
// Host launch
// ---------------------------------------------------------------------------
static inline size_t aln(size_t x) { return (x + 255) & ~(size_t)255; }

extern "C" void kernel_launch(void* const* d_in, const int* in_sizes, int n_in,
                              void* d_out, int out_size, void* d_ws, size_t ws_size,
                              hipStream_t stream) {
  (void)in_sizes; (void)n_in; (void)out_size; (void)ws_size;
  const float* hs      = (const float*)d_in[0];  // [L,H]
  const float* w_in    = (const float*)d_in[1];  // [2*DI,H]
  const float* conv_w  = (const float*)d_in[2];  // [DI,1,4]
  const float* conv_b  = (const float*)d_in[3];  // [DI]
  const float* w_x     = (const float*)d_in[4];  // [96,DI]
  const float* w_dt    = (const float*)d_in[5];  // [DI,DTR]
  const float* dt_b    = (const float*)d_in[6];  // [DI]
  const float* A_log   = (const float*)d_in[7];  // [DI,DS]
  const float* Dv      = (const float*)d_in[8];  // [DI]
  const float* w_out   = (const float*)d_in[9];  // [H,DI]
  float* out = (float*)d_out;                    // [L,H]

  char* ws = (char*)d_ws;
  size_t o = 0;
  h16*   hs_h    = (h16*)  (ws + o); o += aln((size_t)LL * HH * 2);
  h16*   w_in_h  = (h16*)  (ws + o); o += aln((size_t)2 * DI * HH * 2);
  h16*   w_x_h   = (h16*)  (ws + o); o += aln((size_t)NSSM * DI * 2);
  h16*   w_dt_h  = (h16*)  (ws + o); o += aln((size_t)DI * DTR * 2);
  h16*   w_out_h = (h16*)  (ws + o); o += aln((size_t)HH * DI * 2);
  float* proj    = (float*)(ws + o); o += aln((size_t)LL * 2 * DI * 4);
  float* xcf     = (float*)(ws + o); o += aln((size_t)LL * DI * 4);
  h16*   xch     = (h16*)  (ws + o); o += aln((size_t)LL * DI * 2);
  float* ssm     = (float*)(ws + o); o += aln((size_t)LL * NSSM * 4);
  h16*   dtin    = (h16*)  (ws + o); o += aln((size_t)LL * DTR * 2);
  float* dtf     = (float*)(ws + o); o += aln((size_t)LL * DI * 4);
  float* Aneg    = (float*)(ws + o); o += aln((size_t)DI * DS * 4);
  float* yf      = (float*)(ws + o); o += aln((size_t)LL * DI * 4);
  h16*   yg      = (h16*)  (ws + o); o += aln((size_t)LL * DI * 2);

  const int EB = 256;
  auto blocks = [](int n, int b) { return (n + b - 1) / b; };

  // fp32 -> fp16 conversions of activations + weights
  k_f32_to_f16<<<blocks(LL * HH, EB), EB, 0, stream>>>(hs, hs_h, LL * HH);
  k_f32_to_f16<<<blocks(2 * DI * HH, EB), EB, 0, stream>>>(w_in, w_in_h, 2 * DI * HH);
  k_f32_to_f16<<<blocks(NSSM * DI, EB), EB, 0, stream>>>(w_x, w_x_h, NSSM * DI);
  k_f32_to_f16<<<blocks(DI * DTR, EB), EB, 0, stream>>>(w_dt, w_dt_h, DI * DTR);
  k_f32_to_f16<<<blocks(HH * DI, EB), EB, 0, stream>>>(w_out, w_out_h, HH * DI);
  k_negexp<<<blocks(DI * DS, EB), EB, 0, stream>>>(A_log, Aneg, DI * DS);

  // 1) in_proj: proj[L, 2*DI] = hs @ w_in^T           (NT=4, grid 16x64)
  k_gemm_f16<4><<<dim3(LL / 128, (2 * DI) / 64), 128, 0, stream>>>(
      hs_h, w_in_h, proj, 2 * DI, HH);

  // 2) causal depthwise conv + SiLU
  k_conv_silu<<<blocks(LL * DI, EB), EB, 0, stream>>>(proj, conv_w, conv_b, xcf, xch);

  // 3) x_proj: ssm[L, 96] = xc @ w_x^T                (NT=2, grid 16x3)
  k_gemm_f16<2><<<dim3(LL / 128, NSSM / 32), 128, 0, stream>>>(
      xch, w_x_h, ssm, NSSM, DI);

  // 4) dt_proj + softplus                              (NT=4, grid 16x32)
  k_split_dt<<<blocks(LL * DTR, EB), EB, 0, stream>>>(ssm, dtin);
  k_gemm_f16<4><<<dim3(LL / 128, DI / 64), 128, 0, stream>>>(dtin, w_dt_h, dtf, DI, DTR);
  k_dt_softplus<<<blocks(LL * DI, EB), EB, 0, stream>>>(dtf, dt_b);

  // 5) selective scan: 16 lanes per channel, DI*DS = 32768 lanes
  k_scan<<<blocks(DI * DS, EB), EB, 0, stream>>>(dtf, xcf, ssm, Aneg, yf);

  // 6) gate
  k_gate<<<blocks(LL * DI, EB), EB, 0, stream>>>(yf, xcf, Dv, proj, yg);

  // 7) out_proj: out[L, H] = yg @ w_out^T              (NT=4, grid 16x16)
  k_gemm_f16<4><<<dim3(LL / 128, HH / 64), 128, 0, stream>>>(yg, w_out_h, out, HH, DI);
}
